// PrivacyGNN_27212912787886
// MI455X (gfx1250) — matrix operand order repaired and checked by reading, not compiled, not run on hardware
//
#include <hip/hip_runtime.h>
#include <math.h>

// ---------------------------------------------------------------------------
// PrivacyGNN forward for gfx1250 (MI455X).
//  - Dense GEMMs via v_wmma_f32_16x16x32_f16 (wave32, 16x16 tiles, fp32 acc)
//  - Edge scatter/gather via float atomics (node state is L2-resident: 25.6MB)
// ---------------------------------------------------------------------------

typedef __attribute__((ext_vector_type(16))) _Float16 v16h;
typedef __attribute__((ext_vector_type(8)))  float    v8f;

#define HID 64
#define BN_EPS 1e-5f

// ---------------- WMMA fragment loaders (contiguous k order, shared by A and B) ----------------
// A: 16x32 f16 tile of a row-major fp32 matrix. Lane L: M = L&15.
//    half h (0..15): K = k0 + (L>>4)*16 + h
template <bool CONCAT>
__device__ inline v16h load_a_frag(const float* __restrict__ A,
                                   const float* __restrict__ A2,
                                   int lda, int Nn, int row_base, int k0) {
  int lane = threadIdx.x & 31;
  int m = row_base + (lane & 15);
  if (m >= Nn) m = Nn - 1;                 // clamp (keeps EXEC full for WMMA)
  int khalf = lane >> 4;
  v16h a;
#pragma unroll
  for (int h = 0; h < 16; ++h) {
    int k = k0 + khalf * 16 + h;
    float v;
    if (CONCAT) v = (k < 64) ? A[(size_t)m * 64 + k] : A2[(size_t)m * 64 + (k - 64)];
    else        v = A[(size_t)m * lda + k];
    a[h] = (_Float16)v;
  }
  return a;
}

// B: 32x16 f16 tile of row-major fp32 W[K,NOUT]. Lane L: N = n0 + (L&15).
//    half h: K = k0 + (L>>4)*16 + h
__device__ inline v16h load_b_frag(const float* __restrict__ W, int ldb,
                                   int k0, int n0) {
  int lane = threadIdx.x & 31;
  int n = n0 + (lane & 15);
  int kbase = k0 + (lane >> 4) * 16;
  v16h b;
#pragma unroll
  for (int h = 0; h < 16; ++h) b[h] = (_Float16)W[(size_t)(kbase + h) * ldb + n];
  return b;
}

// ---------------- generic WMMA GEMM: C[N,NOUT] = A[N,KDIM] @ W + bias -------
// blockDim = 128 (4 waves), each wave owns a 16-row tile.
template <int KDIM, int NOUT, bool RELU, bool CONCAT>
__global__ void wmma_gemm(const float* __restrict__ A, const float* __restrict__ A2,
                          const float* __restrict__ W, const float* __restrict__ bias,
                          float* __restrict__ C, int Nn) {
  constexpr int NT = NOUT / 16;
  int wave = threadIdx.x >> 5;
  int lane = threadIdx.x & 31;
  int row_base = (blockIdx.x * 4 + wave) * 16;
  if (row_base >= Nn) return;              // uniform per wave

  v8f acc[NT];
#pragma unroll
  for (int t = 0; t < NT; ++t)
#pragma unroll
    for (int r = 0; r < 8; ++r) acc[t][r] = 0.0f;

#pragma unroll
  for (int k0 = 0; k0 < KDIM; k0 += 32) {
    v16h a = load_a_frag<CONCAT>(A, A2, KDIM, Nn, row_base, k0);
#pragma unroll
    for (int t = 0; t < NT; ++t) {
      v16h b = load_b_frag(W, NOUT, k0, t * 16);
      acc[t] = __builtin_amdgcn_wmma_f32_16x16x32_f16(
          /*neg_a=*/false, a, /*neg_b=*/false, b,
          /*c_mod=*/(short)0, acc[t], /*reuse_a=*/false, /*reuse_b=*/false);
    }
  }

  // C/D layout: VGPR r, lanes 0-15 -> M=r, lanes 16-31 -> M=8+r; N = lane&15
  int mrow = row_base + (lane >> 4) * 8;
  int ncol = lane & 15;
#pragma unroll
  for (int t = 0; t < NT; ++t)
#pragma unroll
    for (int r = 0; r < 8; ++r) {
      int m = mrow + r;
      if (m < Nn) {
        int n = t * 16 + ncol;
        float v = acc[t][r] + (bias ? bias[n] : 0.0f);
        if (RELU) v = fmaxf(v, 0.0f);
        C[(size_t)m * NOUT + n] = v;
      }
    }
}

// ---------------- graph kernels ---------------------------------------------
__global__ void init_kernel(float* deg_col, float* deg_row, float* infl,
                            unsigned* scal, int Nn) {
  int i = blockIdx.x * blockDim.x + threadIdx.x;
  if (i < Nn) {
    deg_col[i] = 1.0f;   // self-loop contribution to GCN degree
    deg_row[i] = 0.0f;
    infl[i] = 0.0f;
  }
  if (i == 0) { scal[0] = 0u; scal[1] = 0u; }
}

__global__ void deg_kernel(const int* __restrict__ erow, const int* __restrict__ ecol,
                           float* deg_col, float* deg_row, int E) {
  int e = blockIdx.x * blockDim.x + threadIdx.x;
  if (e >= E) return;
  atomicAdd(&deg_col[ecol[e]], 1.0f);
  atomicAdd(&deg_row[erow[e]], 1.0f);
}

__global__ void infl_kernel(const int* __restrict__ erow, const int* __restrict__ ecol,
                            const float* __restrict__ deg_row, float* infl_sum, int E) {
  int e = blockIdx.x * blockDim.x + threadIdx.x;
  if (e >= E) return;
  atomicAdd(&infl_sum[erow[e]], deg_row[ecol[e]]);
}

__global__ void node_stats(const float* __restrict__ deg_col,
                           const float* __restrict__ deg_row,
                           float* infl, float* dis, unsigned* scal, int Nn) {
  int i = blockIdx.x * blockDim.x + threadIdx.x;
  if (i >= Nn) return;
  dis[i] = rsqrtf(deg_col[i]);                       // deg_col >= 1 always
  float dr = deg_row[i];
  float inf = (dr > 0.0f) ? infl[i] / fmaxf(dr, 1.0f) : 0.0f;
  infl[i] = inf;
  atomicMax(&scal[0], __float_as_uint(dr));          // non-negative -> bit-monotone
  atomicMax(&scal[1], __float_as_uint(inf));
}

// self-loop term: h_acc = hw * dis^2   (also initializes the accumulator)
__global__ void selfloop_kernel(const float* __restrict__ hw,
                                const float* __restrict__ dis,
                                float* __restrict__ hacc, int Nn) {
  int g = blockIdx.x * blockDim.x + threadIdx.x;
  if (g >= Nn * HID) return;
  int i = g >> 6;
  float d = dis[i];
  hacc[g] = hw[g] * d * d;
}

// edge propagation: hacc[col] += hw[row] * dis[row]*dis[col]  (4 cols/thread)
__global__ void edge_prop(const int* __restrict__ erow, const int* __restrict__ ecol,
                          const float* __restrict__ dis, const float* __restrict__ hw,
                          float* __restrict__ hacc, int E) {
  int g = blockIdx.x * blockDim.x + threadIdx.x;
  int e = g >> 4;
  if (e >= E) return;
  int c = (g & 15) << 2;
  int r = erow[e], cc = ecol[e];
  float nrm = dis[r] * dis[cc];
  const float4 v = *(const float4*)(hw + (size_t)r * HID + c);
  float* dst = hacc + (size_t)cc * HID + c;
  atomicAdd(dst + 0, v.x * nrm);
  atomicAdd(dst + 1, v.y * nrm);
  atomicAdd(dst + 2, v.z * nrm);
  atomicAdd(dst + 3, v.w * nrm);
}

// h = relu((hacc + b) * gamma/sqrt(1+eps) + beta)
__global__ void bn_relu_kernel(const float* __restrict__ hacc,
                               const float* __restrict__ b_gcn,
                               const float* __restrict__ gamma,
                               const float* __restrict__ beta,
                               float* __restrict__ h, int layer, int Nn) {
  int g = blockIdx.x * blockDim.x + threadIdx.x;
  if (g >= Nn * HID) return;
  int c = g & 63;
  float scale = gamma[layer * HID + c] * rsqrtf(1.0f + BN_EPS);
  float v = (hacc[g] + b_gcn[layer * HID + c]) * scale + beta[layer * HID + c];
  h[g] = fmaxf(v, 0.0f);
}

// structural features + tiny MLP (3 -> 32 -> 64), one thread per node
__global__ void struct_mlp(const float* __restrict__ deg_row,
                           const float* __restrict__ infl,
                           const unsigned* __restrict__ scal,
                           const float* __restrict__ W_s1, const float* __restrict__ b_s1,
                           const float* __restrict__ W_s2, const float* __restrict__ b_s2,
                           float* __restrict__ se, int Nn) {
  int i = blockIdx.x * blockDim.x + threadIdx.x;
  if (i >= Nn) return;
  float dmax = __uint_as_float(scal[0]);
  float imax = __uint_as_float(scal[1]);
  float nd = deg_row[i];
  nd = (dmax > 0.0f) ? nd / dmax : nd;
  float inf = infl[i];
  inf = (imax > 0.0f) ? inf / imax : inf;
  // sf = [nd, 0 (clustering), inf]
  float hmid[32];
#pragma unroll
  for (int j = 0; j < 32; ++j) {
    float s = b_s1[j] + nd * W_s1[0 * 32 + j] + inf * W_s1[2 * 32 + j];
    hmid[j] = fmaxf(s, 0.0f);
  }
  float* out = se + (size_t)i * HID;
  for (int c = 0; c < HID; ++c) {
    float s = b_s2[c];
#pragma unroll
    for (int j = 0; j < 32; ++j) s += hmid[j] * W_s2[j * HID + c];
    out[c] = s;
  }
}

// out = sigmoid(o2 @ W_o3 + b_o3), o2: [N,32]
__global__ void head_kernel(const float* __restrict__ o2,
                            const float* __restrict__ W_o3,
                            const float* __restrict__ b_o3,
                            float* __restrict__ out, int Nn) {
  int i = blockIdx.x * blockDim.x + threadIdx.x;
  if (i >= Nn) return;
  const float* v = o2 + (size_t)i * 32;
  float s = b_o3[0];
#pragma unroll
  for (int k = 0; k < 32; ++k) s += v[k] * W_o3[k];
  out[i] = 1.0f / (1.0f + expf(-s));
}

// ---------------------------------------------------------------------------
extern "C" void kernel_launch(void* const* d_in, const int* in_sizes, int n_in,
                              void* d_out, int out_size, void* d_ws, size_t ws_size,
                              hipStream_t stream) {
  const float* x      = (const float*)d_in[0];
  const int*   ei     = (const int*)  d_in[1];
  const float* W_in   = (const float*)d_in[2];
  const float* b_in   = (const float*)d_in[3];
  const float* W_gcn  = (const float*)d_in[4];
  const float* b_gcn  = (const float*)d_in[5];
  const float* gamma  = (const float*)d_in[6];
  const float* beta   = (const float*)d_in[7];
  const float* W_s1   = (const float*)d_in[8];
  const float* b_s1   = (const float*)d_in[9];
  const float* W_s2   = (const float*)d_in[10];
  const float* b_s2   = (const float*)d_in[11];
  const float* W_o1   = (const float*)d_in[12];
  const float* b_o1   = (const float*)d_in[13];
  const float* W_o2   = (const float*)d_in[14];
  const float* b_o2   = (const float*)d_in[15];
  const float* W_o3   = (const float*)d_in[16];
  const float* b_o3   = (const float*)d_in[17];

  const int N = in_sizes[0] / 128;
  const int E = in_sizes[1] / 2;
  const int* erow = ei;
  const int* ecol = ei + E;

  // ---- workspace carve-up (~104 MB for N=100k) ----
  float* ws = (float*)d_ws;
  float* h    = ws; ws += (size_t)N * HID;
  float* hw   = ws; ws += (size_t)N * HID;   // reused as o1 after GCN layers
  float* hacc = ws; ws += (size_t)N * HID;   // reused as o2 after GCN layers
  float* se   = ws; ws += (size_t)N * HID;
  float* degc = ws; ws += N;
  float* degr = ws; ws += N;
  float* infl = ws; ws += N;
  float* dis  = ws; ws += N;
  unsigned* scal = (unsigned*)ws;
  float* o1 = hw;
  float* o2 = hacc;

  const int TB = 256;
  dim3 blk(TB);
  dim3 gN((N + TB - 1) / TB);
  dim3 gE((E + TB - 1) / TB);
  dim3 gE16(((size_t)E * 16 + TB - 1) / TB);
  dim3 gNH(((size_t)N * HID + TB - 1) / TB);
  dim3 gemm_grid((N + 63) / 64);
  dim3 gemm_blk(128);

  // ---- graph statistics ----
  init_kernel<<<gN, blk, 0, stream>>>(degc, degr, infl, scal, N);
  deg_kernel<<<gE, blk, 0, stream>>>(erow, ecol, degc, degr, E);
  infl_kernel<<<gE, blk, 0, stream>>>(erow, ecol, degr, infl, E);
  node_stats<<<gN, blk, 0, stream>>>(degc, degr, infl, dis, scal, N);

  // ---- h0 = x @ W_in + b_in   (WMMA, K=128 -> 64) ----
  wmma_gemm<128, 64, false, false><<<gemm_grid, gemm_blk, 0, stream>>>(
      x, nullptr, W_in, b_in, h, N);

  // ---- 3 GCN layers ----
  for (int l = 0; l < 3; ++l) {
    wmma_gemm<64, 64, false, false><<<gemm_grid, gemm_blk, 0, stream>>>(
        h, nullptr, W_gcn + (size_t)l * HID * HID, nullptr, hw, N);
    selfloop_kernel<<<gNH, blk, 0, stream>>>(hw, dis, hacc, N);
    edge_prop<<<gE16, blk, 0, stream>>>(erow, ecol, dis, hw, hacc, E);
    bn_relu_kernel<<<gNH, blk, 0, stream>>>(hacc, b_gcn, gamma, beta, h, l, N);
  }

  // ---- structural embedding ----
  struct_mlp<<<gN, blk, 0, stream>>>(degr, infl, scal, W_s1, b_s1, W_s2, b_s2, se, N);

  // ---- output MLP ----
  wmma_gemm<128, 64, true, true><<<gemm_grid, gemm_blk, 0, stream>>>(
      h, se, W_o1, b_o1, o1, N);
  wmma_gemm<64, 32, true, false><<<gemm_grid, gemm_blk, 0, stream>>>(
      o1, nullptr, W_o2, b_o2, o2, N);
  head_kernel<<<gN, blk, 0, stream>>>(o2, W_o3, b_o3, (float*)d_out, N);
}
